// LeAttention_57131654971958
// MI455X (gfx1250) — compile-verified, matrix operand order checked by reading, not run on hardware
//
#include <hip/hip_runtime.h>
#include <cstddef>

typedef __bf16          v16bf  __attribute__((ext_vector_type(16)));
typedef unsigned short  v16u16 __attribute__((ext_vector_type(16)));
typedef float           v8f    __attribute__((ext_vector_type(8)));

#define EPS_      1e-5f
#define SCALE_    0.17677669529663687f   /* 32^-0.5 */
#define INVSCALE_ 5.656854249492381f     /* 32^0.5  */

__device__ __forceinline__ unsigned short f2bf(float f) {
  unsigned int u = __float_as_uint(f);
  u += 0x7FFFu + ((u >> 16) & 1u);            // round-to-nearest-even
  return (unsigned short)(u >> 16);
}

union V16 { v16u16 u; v16bf b; unsigned short h[16]; };

__device__ __forceinline__ v16bf ldb(const unsigned short* p) {
  V16 v; v.u = *reinterpret_cast<const v16u16*>(p); return v.b;
}

__device__ __forceinline__ v8f wmma_bf16(v16bf a, v16bf b, v8f c) {
  return __builtin_amdgcn_wmma_f32_16x16x32_bf16(false, a, false, b, (short)0, c, false, false);
}

// ---------------- 1) fold BN into QKV weights -> Wcat[1024][256] bf16 -------
__global__ void k_fold_qkv(const float* Wq, const float* qg, const float* qb, const float* qm, const float* qv,
                           const float* Wk, const float* kg, const float* kb, const float* km, const float* kv,
                           const float* Wv, const float* vg, const float* vb, const float* vm, const float* vv,
                           unsigned short* Wcat, float* beta) {
  int o = blockIdx.x;            // 0..1023  (0-255 q, 256-511 k, 512-1023 v)
  int c = threadIdx.x;           // 0..255
  const float* W; float g, bb, mm, vv_; int oo;
  if (o < 256)      { oo = o;       W = Wq; g = qg[oo]; bb = qb[oo]; mm = qm[oo]; vv_ = qv[oo]; }
  else if (o < 512) { oo = o - 256; W = Wk; g = kg[oo]; bb = kb[oo]; mm = km[oo]; vv_ = kv[oo]; }
  else              { oo = o - 512; W = Wv; g = vg[oo]; bb = vb[oo]; mm = vm[oo]; vv_ = vv[oo]; }
  float inv = g * rsqrtf(vv_ + EPS_);
  Wcat[(size_t)o * 256 + c] = f2bf(W[(size_t)oo * 256 + c] * inv);
  if (c == 0) beta[o] = bb - mm * inv;
}

// ---------------- fold bo + output BN into Wo -> WoB[256][512] bf16 ---------
__global__ void k_fold_wo(const float* Wo, const float* bo, const float* og, const float* ob,
                          const float* om, const float* ov, unsigned short* WoB, float* betaO) {
  int o = blockIdx.x; int t = threadIdx.x;   // 256 blocks x 256 threads
  float inv = og[o] * rsqrtf(ov[o] + EPS_);
  WoB[(size_t)o * 512 + t]       = f2bf(Wo[(size_t)o * 512 + t] * inv);
  WoB[(size_t)o * 512 + t + 256] = f2bf(Wo[(size_t)o * 512 + t + 256] * inv);
  if (t == 0) betaO[o] = bo[o] * inv + ob[o] - om[o] * inv;
}

// ---------------- 2) transpose x[b][c][p] -> xT[b][p][c] bf16 ---------------
__global__ void k_transpose_x(const float* x, unsigned short* xT) {
  __shared__ float T[64][65];
  int pblk = blockIdx.x, cblk = blockIdx.y, b = blockIdx.z;
  int tid = threadIdx.x;
  #pragma unroll
  for (int it = 0; it < 16; ++it) {
    int idx = it * 256 + tid;
    int ci = idx >> 6, pj = idx & 63;
    T[ci][pj] = x[((size_t)(b * 256 + cblk * 64 + ci)) * 1024 + pblk * 64 + pj];
  }
  __syncthreads();
  #pragma unroll
  for (int it = 0; it < 16; ++it) {
    int idx = it * 256 + tid;
    int pi = idx >> 6, cj = idx & 63;
    xT[((size_t)(b * 1024 + pblk * 64 + pi)) * 256 + cblk * 64 + cj] = f2bf(T[cj][pi]);
  }
}

// ---------------- 3) QKV projection GEMM (bf16 WMMA, BN folded) -------------
// qT,kT: [b*8+n][s][32]  (transposed, A/B friendly for QK^T)
// vN:    [b][o=512][s]   (natural, B friendly for PV)
__global__ void k_qkv(const unsigned short* xT, const unsigned short* Wcat, const float* beta,
                      unsigned short* qT, unsigned short* kT, unsigned short* vN) {
  int ptile = blockIdx.x, nchunk = blockIdx.y, b = blockIdx.z;
  int w = threadIdx.x >> 5, lid = threadIdx.x & 31;
  int l16 = lid & 15, hi = lid >> 4;
  int msub = w & 3, nh = w >> 2;
  int nbase = nchunk * 128 + nh * 64;
  int prow = b * 1024 + ptile * 64 + msub * 16 + l16;
  const v8f vz = {0.f,0.f,0.f,0.f,0.f,0.f,0.f,0.f};
  v8f acc[4] = {vz, vz, vz, vz};
  for (int kk = 0; kk < 8; ++kk) {              // K = 256 / 32
    v16bf a = ldb(xT + (size_t)prow * 256 + kk * 32 + hi * 16);
    #pragma unroll
    for (int t = 0; t < 4; ++t) {
      int o = nbase + t * 16 + l16;
      v16bf bt = ldb(Wcat + (size_t)o * 256 + kk * 32 + hi * 16);
      acc[t] = wmma_bf16(a, bt, acc[t]);
    }
  }
  #pragma unroll
  for (int t = 0; t < 4; ++t) {
    int o = nbase + t * 16 + l16;
    float bta = beta[o];
    #pragma unroll
    for (int j = 0; j < 8; ++j) {
      int M = j + hi * 8;
      int p = ptile * 64 + msub * 16 + M;
      unsigned short hval = f2bf(acc[t][j] + bta);
      if (nchunk < 2) {                                   // q (uniform per block)
        int n = o >> 5, d = o & 31;
        qT[(((size_t)(b * 8 + n)) * 1024 + p) * 32 + d] = hval;
      } else if (nchunk < 4) {                            // k
        int o2 = o - 256, n = o2 >> 5, d = o2 & 31;
        kT[(((size_t)(b * 8 + n)) * 1024 + p) * 32 + d] = hval;
      } else {                                            // v
        int o2 = o - 512;
        vN[((size_t)(b * 512 + o2)) * 1024 + p] = hval;
      }
    }
  }
}

// ---------------- 4) fused attention + GELU ---------------------------------
// 2 waves/WG, each wave: one (b,n,16-query) tile. Scores in LDS fp32.
// Row max tracked in registers during QK^T; exp stored back during sum scan.
__global__ void k_attn(const unsigned short* qT, const unsigned short* kT, const unsigned short* vN,
                       const float* posemb, unsigned short* goT) {
  extern __shared__ float smem[];
  float* poscol = smem;                    // 1024 floats: pos_emb[:, n]
  float* rmaxs  = smem + 1024;             // 2 waves * 16 row maxima
  float* sc     = smem + 1056;             // 2 waves * 16 rows * 1032 floats
  int qgrp = blockIdx.x, n = blockIdx.y, b = blockIdx.z;
  int tid = threadIdx.x;
  int w = tid >> 5, lid = tid & 31, l16 = lid & 15, hi = lid >> 4;
  for (int k = tid; k < 1024; k += 64) poscol[k] = posemb[k * 8 + n];
  __syncthreads();

  int qbase = (qgrp * 2 + w) * 16;
  int bn = b * 8 + n;
  v16bf aq = ldb(qT + (((size_t)bn) * 1024 + qbase + l16) * 32 + hi * 16);
  float* myS = sc + w * 16 * 1032;
  const v8f vz = {0.f,0.f,0.f,0.f,0.f,0.f,0.f,0.f};

  // pass 1: S = q.k * SCALE + bias / SCALE -> LDS; running per-row max in regs
  float pmax[8];
  #pragma unroll
  for (int j = 0; j < 8; ++j) pmax[j] = -1e30f;
  for (int kt = 0; kt < 64; ++kt) {
    v16bf bk = ldb(kT + (((size_t)bn) * 1024 + kt * 16 + l16) * 32 + hi * 16);
    v8f s = wmma_bf16(aq, bk, vz);
    int kgl = kt * 16 + l16;
    int kr = kgl >> 5, kc = kgl & 31;
    #pragma unroll
    for (int j = 0; j < 8; ++j) {
      int M = j + hi * 8;
      int qgl = qbase + M, qr = qgl >> 5, qc = qgl & 31;
      int dr = qr - kr; if (dr < 0) dr = -dr;
      int dc = qc - kc; if (dc < 0) dc = -dc;
      float val = s[j] * SCALE_ + poscol[dr * 32 + dc] * INVSCALE_;
      myS[M * 1032 + kgl] = val;
      pmax[j] = fmaxf(pmax[j], val);
    }
  }
  // butterfly over the 16 lanes of each half (half hi owns rows hi*8..hi*8+7;
  // its 16 lanes jointly cover all 1024 columns) -> exact row maxima
  #pragma unroll
  for (int j = 0; j < 8; ++j) {
    float m = pmax[j];
    m = fmaxf(m, __shfl_xor(m, 1));
    m = fmaxf(m, __shfl_xor(m, 2));
    m = fmaxf(m, __shfl_xor(m, 4));
    m = fmaxf(m, __shfl_xor(m, 8));
    if (l16 == j) rmaxs[w * 16 + j + 8 * hi] = m;
  }
  __syncthreads();

  // sum scan: lane handles row l16, half hi; exp stored back in-place
  float mx = rmaxs[w * 16 + l16];
  float4* rowp = reinterpret_cast<float4*>(myS + l16 * 1032 + hi * 512);
  float sum = 0.f;
  for (int c4 = 0; c4 < 128; ++c4) {
    float4 f = rowp[c4];
    f.x = __expf(f.x - mx);
    f.y = __expf(f.y - mx);
    f.z = __expf(f.z - mx);
    f.w = __expf(f.w - mx);
    sum += (f.x + f.y) + (f.z + f.w);
    rowp[c4] = f;
  }
  sum += __shfl_xor(sum, 16);
  float rinv = 1.f / sum;
  __syncthreads();

  // pass 2: P (normalized, bf16, A-layout) @ V  -> O[16 x 64]
  v8f oacc[4] = {vz, vz, vz, vz};
  for (int kc4 = 0; kc4 < 32; ++kc4) {
    V16 ap;
    int ks = kc4 * 32 + hi * 16;
    #pragma unroll
    for (int q4 = 0; q4 < 4; ++q4) {
      float4 f = *reinterpret_cast<const float4*>(myS + l16 * 1032 + ks + q4 * 4);
      ap.h[q4 * 4 + 0] = f2bf(f.x * rinv);
      ap.h[q4 * 4 + 1] = f2bf(f.y * rinv);
      ap.h[q4 * 4 + 2] = f2bf(f.z * rinv);
      ap.h[q4 * 4 + 3] = f2bf(f.w * rinv);
    }
    #pragma unroll
    for (int t = 0; t < 4; ++t) {
      v16bf bv = ldb(vN + ((size_t)(b * 512 + n * 64 + t * 16 + l16)) * 1024 + kc4 * 32 + hi * 16);
      oacc[t] = wmma_bf16(ap.b, bv, oacc[t]);
    }
  }

  // GELU(erf) then store pixel-major bf16 for the output GEMM
  #pragma unroll
  for (int t = 0; t < 4; ++t) {
    #pragma unroll
    for (int j = 0; j < 8; ++j) {
      int M = j + hi * 8;
      float xv = oacc[t][j];
      float gl = 0.5f * xv * (1.f + erff(xv * 0.70710678118654752f));
      goT[((size_t)(b * 1024 + qbase + M)) * 512 + n * 64 + t * 16 + l16] = f2bf(gl);
    }
  }
}

// ---------------- 5) output projection GEMM ---------------------------------
__global__ void k_out(const unsigned short* goT, const unsigned short* WoB, const float* betaO, float* out) {
  int ptile = blockIdx.x, nchunk = blockIdx.y, b = blockIdx.z;
  int w = threadIdx.x >> 5, lid = threadIdx.x & 31, l16 = lid & 15, hi = lid >> 4;
  int msub = w & 3, nh = w >> 2;
  int nbase = nchunk * 128 + nh * 64;
  int prow = b * 1024 + ptile * 64 + msub * 16 + l16;
  const v8f vz = {0.f,0.f,0.f,0.f,0.f,0.f,0.f,0.f};
  v8f acc[4] = {vz, vz, vz, vz};
  for (int kk = 0; kk < 16; ++kk) {            // K = 512 / 32
    v16bf a = ldb(goT + (size_t)prow * 512 + kk * 32 + hi * 16);
    #pragma unroll
    for (int t = 0; t < 4; ++t) {
      int o = nbase + t * 16 + l16;
      v16bf bt = ldb(WoB + (size_t)o * 512 + kk * 32 + hi * 16);
      acc[t] = wmma_bf16(a, bt, acc[t]);
    }
  }
  #pragma unroll
  for (int t = 0; t < 4; ++t) {
    int o = nbase + t * 16 + l16;
    float bta = betaO[o];
    #pragma unroll
    for (int j = 0; j < 8; ++j) {
      int M = j + hi * 8;
      int p = ptile * 64 + msub * 16 + M;
      out[((size_t)(b * 256 + o)) * 1024 + p] = acc[t][j] + bta;
    }
  }
}

// ---------------- launch -----------------------------------------------------
extern "C" void kernel_launch(void* const* d_in, const int* in_sizes, int n_in,
                              void* d_out, int out_size, void* d_ws, size_t ws_size,
                              hipStream_t stream) {
  (void)in_sizes; (void)n_in; (void)out_size; (void)ws_size;
  const float* x    = (const float*)d_in[0];
  const float* Wq   = (const float*)d_in[1];
  const float* q_g  = (const float*)d_in[2];
  const float* q_b  = (const float*)d_in[3];
  const float* q_m  = (const float*)d_in[4];
  const float* q_v  = (const float*)d_in[5];
  const float* Wk   = (const float*)d_in[6];
  const float* k_g  = (const float*)d_in[7];
  const float* k_b  = (const float*)d_in[8];
  const float* k_m  = (const float*)d_in[9];
  const float* k_v  = (const float*)d_in[10];
  const float* Wv   = (const float*)d_in[11];
  const float* v_g  = (const float*)d_in[12];
  const float* v_b  = (const float*)d_in[13];
  const float* v_m  = (const float*)d_in[14];
  const float* v_v  = (const float*)d_in[15];
  const float* pose = (const float*)d_in[16];
  const float* Wo   = (const float*)d_in[17];
  const float* bo   = (const float*)d_in[18];
  const float* o_g  = (const float*)d_in[19];
  const float* o_b  = (const float*)d_in[20];
  const float* o_m  = (const float*)d_in[21];
  const float* o_v  = (const float*)d_in[22];
  // d_in[23] = pos_indices (int32) -- recomputed arithmetically in-kernel

  char* ws = (char*)d_ws;
  unsigned short* Wcat  = (unsigned short*)(ws + 0);          // 1024*256*2 = 512 KB
  float*          beta  = (float*)         (ws + 524288);     // 4 KB
  unsigned short* WoB   = (unsigned short*)(ws + 528384);     // 256*512*2 = 256 KB
  float*          betaO = (float*)         (ws + 790528);     // 4 KB
  unsigned short* xT    = (unsigned short*)(ws + 794624);     // 8 MB
  unsigned short* qT    = (unsigned short*)(ws + 9183232);    // 8 MB
  unsigned short* kT    = (unsigned short*)(ws + 17571840);   // 8 MB
  unsigned short* vN    = (unsigned short*)(ws + 25960448);   // 16 MB
  unsigned short* goT   = (unsigned short*)(ws + 42737664);   // 16 MB -> total ~59.5 MB

  k_fold_qkv<<<dim3(1024), dim3(256), 0, stream>>>(Wq, q_g, q_b, q_m, q_v,
                                                   Wk, k_g, k_b, k_m, k_v,
                                                   Wv, v_g, v_b, v_m, v_v, Wcat, beta);
  k_fold_wo<<<dim3(256), dim3(256), 0, stream>>>(Wo, bo, o_g, o_b, o_m, o_v, WoB, betaO);
  k_transpose_x<<<dim3(16, 4, 16), dim3(256), 0, stream>>>(x, xT);
  k_qkv<<<dim3(16, 8, 16), dim3(256), 0, stream>>>(xT, Wcat, beta, qT, kT, vN);
  size_t smem = (size_t)(1056 + 2 * 16 * 1032) * sizeof(float);   // ~136 KB of 320 KB LDS
  k_attn<<<dim3(32, 8, 16), dim3(64), smem, stream>>>(qT, kT, vN, pose, goT);
  k_out<<<dim3(16, 2, 16), dim3(256), 0, stream>>>(goT, WoB, betaO, (float*)d_out);
}
